// MixtureOfExperts_90125593739686
// MI455X (gfx1250) — compile-verified, matrix operand order checked by reading, not run on hardware
//
#include <hip/hip_runtime.h>
#include <hip/hip_bf16.h>
#include <math.h>

#define T_TOK 8192
#define DIM   1024
#define FFD   4096
#define NEXP  8

typedef __bf16 bf16_t;
typedef __attribute__((ext_vector_type(16))) __bf16 v16bf;
typedef __attribute__((ext_vector_type(8)))  __bf16 v8bf;
typedef __attribute__((ext_vector_type(4)))  __bf16 v4bf;
typedef __attribute__((ext_vector_type(2)))  __bf16 v2bf;
typedef __attribute__((ext_vector_type(8)))  float  v8f;

// ---------------- LDS layout (bytes) for FFN kernel ----------------
// xA : 64 rows x 1032 bf16 (stride pad)            = 132096
// B1 : 2 buffers x 64 f x 72 bf16                  =  18432
// hA : 64 rows x 72 bf16                           =   9216
// B2 : 1024 n x 72 bf16                            = 147456
#define XA_OFF 0
#define XA_S   1032
#define B1_OFF 132096
#define B1_S   72
#define HA_OFF 150528
#define HA_S   72
#define B2_OFF 159744
#define B2_S   72
#define LDS_BYTES 307200

__device__ inline v16bf load_frag(const bf16_t* base, int strideElems, int row, int k0) {
    const bf16_t* p = base + row * strideElems + k0;
    v8bf lo = *(const v8bf*)(p);
    v8bf hi = *(const v8bf*)(p + 16);
    v16bf r;
#pragma unroll
    for (int i = 0; i < 8; ++i) { r[i] = lo[i]; r[i + 8] = hi[i]; }
    return r;
}

// ---------------- Kernel 0: zero output + control region ----------------
__global__ void moe_zero_kernel(float* __restrict__ out, int n, unsigned int* __restrict__ ctl) {
    int i = blockIdx.x * blockDim.x + threadIdx.x;
    if (i < n) out[i] = 0.0f;
    if (blockIdx.x == 0 && threadIdx.x < 64) ctl[threadIdx.x] = 0u;
}

// ---------------- Kernel 1: gating (one wave32 per token) ----------------
__global__ void moe_gate_kernel(const float* __restrict__ x, const float* __restrict__ gW,
                                const float* __restrict__ gb,
                                int* __restrict__ e0, int* __restrict__ e1,
                                float* __restrict__ w0, float* __restrict__ w1,
                                int* __restrict__ cnt, float* __restrict__ ssum) {
    int t = blockIdx.x * 8 + (threadIdx.x >> 5);
    int lane = threadIdx.x & 31;
    float acc[NEXP];
#pragma unroll
    for (int e = 0; e < NEXP; ++e) acc[e] = 0.0f;
    const float* xr = x + (size_t)t * DIM;
    for (int d = lane; d < DIM; d += 32) {
        float xv = xr[d];
        const float4* g4 = (const float4*)(gW + (size_t)d * NEXP);
        float4 a = g4[0], b = g4[1];
        acc[0] += xv * a.x; acc[1] += xv * a.y; acc[2] += xv * a.z; acc[3] += xv * a.w;
        acc[4] += xv * b.x; acc[5] += xv * b.y; acc[6] += xv * b.z; acc[7] += xv * b.w;
    }
#pragma unroll
    for (int off = 16; off > 0; off >>= 1)
#pragma unroll
        for (int e = 0; e < NEXP; ++e) acc[e] += __shfl_down(acc[e], off, 32);
    if (lane == 0) {
        float sc[NEXP];
        float mx = -1e30f;
#pragma unroll
        for (int e = 0; e < NEXP; ++e) { acc[e] += gb[e]; mx = fmaxf(mx, acc[e]); }
        float sum = 0.0f;
#pragma unroll
        for (int e = 0; e < NEXP; ++e) { sc[e] = __expf(acc[e] - mx); sum += sc[e]; }
        float inv = 1.0f / sum;
        int i0 = 0;
#pragma unroll
        for (int e = 1; e < NEXP; ++e) if (sc[e] > sc[i0]) i0 = e;
        int i1 = (i0 == 0) ? 1 : 0;
#pragma unroll
        for (int e = 0; e < NEXP; ++e) if (e != i0 && sc[e] > sc[i1]) i1 = e;
        float s0 = sc[i0] * inv, s1 = sc[i1] * inv;
        float rn = 1.0f / (s0 + s1);
        e0[t] = i0; e1[t] = i1; w0[t] = s0 * rn; w1[t] = s1 * rn;
        atomicAdd(cnt + i0, 1);
        atomicAdd(cnt + i1, 1);
#pragma unroll
        for (int e = 0; e < NEXP; ++e) atomicAdd(ssum + e, sc[e] * inv);
    }
}

// ---------------- Kernel 2: scatter tokens into padded per-expert lists ----------------
__global__ void moe_scatter_kernel(const int* __restrict__ e0, const int* __restrict__ e1,
                                   const float* __restrict__ w0, const float* __restrict__ w1,
                                   int* __restrict__ cur, int* __restrict__ pairTok,
                                   float* __restrict__ pairW) {
    int t = blockIdx.x * blockDim.x + threadIdx.x;
    if (t >= T_TOK) return;
    int a = e0[t];
    int p = atomicAdd(cur + a, 1);
    pairTok[(size_t)a * T_TOK + p] = t;
    pairW[(size_t)a * T_TOK + p] = w0[t];
    int b = e1[t];
    int q = atomicAdd(cur + b, 1);
    pairTok[(size_t)b * T_TOK + q] = t;
    pairW[(size_t)b * T_TOK + q] = w1[t];
}

// ---------------- Kernel 3: aux loss scalar ----------------
__global__ void moe_aux_kernel(const int* __restrict__ cnt, const float* __restrict__ ssum,
                               float* __restrict__ out_aux) {
    if (threadIdx.x == 0) {
        float aux = 0.0f;
#pragma unroll
        for (int e = 0; e < NEXP; ++e) {
            float frac = (float)cnt[e] / (float)(T_TOK * 2);
            float rw = ssum[e] / (float)T_TOK;
            aux += frac * rw;
        }
        out_aux[0] = (float)NEXP * aux;
    }
}

// ---------------- Kernel 4: fused expert FFN (bf16 WMMA) ----------------
__global__ __launch_bounds__(512)
void moe_ffn_kernel(const float* __restrict__ x, const float* __restrict__ W1,
                    const float* __restrict__ b1, const float* __restrict__ W2,
                    const float* __restrict__ b2, const int* __restrict__ pairTok,
                    const float* __restrict__ pairW, const int* __restrict__ cnt,
                    float* __restrict__ out) {
    extern __shared__ char smem[];
    bf16_t* xA = (bf16_t*)(smem + XA_OFF);
    bf16_t* B1 = (bf16_t*)(smem + B1_OFF);
    bf16_t* hA = (bf16_t*)(smem + HA_OFF);
    bf16_t* B2 = (bf16_t*)(smem + B2_OFF);

    const int e = blockIdx.y;
    const int tileBase = blockIdx.x * 64;
    const int c = cnt[e];
    if (tileBase >= c) return;
    const int rowsValid = min(64, c - tileBase);

    const int tid = threadIdx.x;
    const int w   = tid >> 5;
    const int l   = tid & 31;
    const int l16 = l & 15;
    const int lhi = l >> 4;
    const int mi  = w >> 2;
    const int ni  = w & 3;

    const int*   tokList = pairTok + (size_t)e * T_TOK + tileBase;
    const float* wList   = pairW   + (size_t)e * T_TOK + tileBase;

    // Gather 64 token rows (fp32 -> bf16) into LDS, packed b64 stores
#pragma unroll 4
    for (int it = 0; it < 32; ++it) {
        int li  = it * 512 + tid;
        int row = li >> 8;
        int c4  = (li & 255) * 4;
        float4 v = make_float4(0.f, 0.f, 0.f, 0.f);
        if (row < rowsValid) {
            int tok = tokList[row];
            v = *(const float4*)(x + (size_t)tok * DIM + c4);
        }
        v4bf p;
        p[0] = (bf16_t)v.x; p[1] = (bf16_t)v.y; p[2] = (bf16_t)v.z; p[3] = (bf16_t)v.w;
        *(v4bf*)(xA + row * XA_S + c4) = p;
    }

    v8f C2[16];
#pragma unroll
    for (int i = 0; i < 16; ++i)
#pragma unroll
        for (int j = 0; j < 8; ++j) C2[i][j] = 0.0f;

    for (int ch = 0; ch < 64; ++ch) {
        const int fb = ch * 64;
        __syncthreads();  // prev chunk's B2 reads done; xA ready on first iter

        // Prefetch next W2 chunk row starts
        if (ch + 1 < 64 && tid < 64)
            __builtin_prefetch(W2 + ((size_t)e * FFD + fb + 64 + tid) * DIM, 0, 0);

        // Stage W2 chunk [64 k x 1024 n] -> B2[n][k] (bf16, transposed, packed k-pairs)
#pragma unroll 4
        for (int it = 0; it < 16; ++it) {
            int li = it * 512 + tid;          // 0..8191
            int k2 = (li >> 8) * 2;           // even k
            int n4 = (li & 255) * 4;
            const float* w2p = W2 + ((size_t)e * FFD + fb + k2) * DIM + n4;
            float4 va = *(const float4*)(w2p);
            float4 vb = *(const float4*)(w2p + DIM);
            v2bf p0; p0[0] = (bf16_t)va.x; p0[1] = (bf16_t)vb.x;
            v2bf p1; p1[0] = (bf16_t)va.y; p1[1] = (bf16_t)vb.y;
            v2bf p2; p2[0] = (bf16_t)va.z; p2[1] = (bf16_t)vb.z;
            v2bf p3; p3[0] = (bf16_t)va.w; p3[1] = (bf16_t)vb.w;
            *(v2bf*)(B2 + (n4 + 0) * B2_S + k2) = p0;
            *(v2bf*)(B2 + (n4 + 1) * B2_S + k2) = p1;
            *(v2bf*)(B2 + (n4 + 2) * B2_S + k2) = p2;
            *(v2bf*)(B2 + (n4 + 3) * B2_S + k2) = p3;
        }

        // -------- Stage 1: h_tile = x[64,1024] @ W1[:, fb:fb+64] --------
        v8f C1;
#pragma unroll
        for (int j = 0; j < 8; ++j) C1[j] = 0.0f;

        for (int st = 0; st < 16; ++st) {     // 16 stages of 64 K, double-buffered
            const int kb = st * 64;
            bf16_t* b1buf = B1 + (st & 1) * (64 * B1_S);
            {   // stage W1 block [64 k x 64 f] -> b1buf[f][k], packed k-pairs
                int k2 = (tid >> 4) * 2;      // even k, 0..62
                int f4 = (tid & 15) * 4;
                const float* w1p = W1 + ((size_t)e * DIM + kb + k2) * FFD + fb + f4;
                float4 va = *(const float4*)(w1p);
                float4 vb = *(const float4*)(w1p + FFD);
                v2bf p0; p0[0] = (bf16_t)va.x; p0[1] = (bf16_t)vb.x;
                v2bf p1; p1[0] = (bf16_t)va.y; p1[1] = (bf16_t)vb.y;
                v2bf p2; p2[0] = (bf16_t)va.z; p2[1] = (bf16_t)vb.z;
                v2bf p3; p3[0] = (bf16_t)va.w; p3[1] = (bf16_t)vb.w;
                *(v2bf*)(b1buf + (f4 + 0) * B1_S + k2) = p0;
                *(v2bf*)(b1buf + (f4 + 1) * B1_S + k2) = p1;
                *(v2bf*)(b1buf + (f4 + 2) * B1_S + k2) = p2;
                *(v2bf*)(b1buf + (f4 + 3) * B1_S + k2) = p3;
            }
            __syncthreads();
#pragma unroll
            for (int ks = 0; ks < 2; ++ks) {
                int k0 = ks * 32 + lhi * 8;
                v16bf a = load_frag(xA, XA_S, mi * 16 + l16, kb + k0);
                v16bf b = load_frag(b1buf, B1_S, ni * 16 + l16, k0);
                C1 = __builtin_amdgcn_wmma_f32_16x16x32_bf16(false, a, false, b, (short)0,
                                                             C1, false, false);
            }
        }

        // bias + exact GELU -> hA (bf16)
        {
            int fcol = ni * 16 + l16;
            float bias = b1[(size_t)e * FFD + fb + fcol];
#pragma unroll
            for (int r = 0; r < 8; ++r) {
                float v = C1[r] + bias;
                float g = 0.5f * v * (1.0f + erff(v * 0.70710678118654752f));
                int m = mi * 16 + lhi * 8 + r;
                hA[m * HA_S + fcol] = (bf16_t)g;
            }
        }
        __syncthreads();  // hA + B2 staging complete

        // -------- Stage 2: y[64,1024] += h[64,64] @ W2[fb:fb+64, :] --------
#pragma unroll
        for (int ks2 = 0; ks2 < 2; ++ks2) {
            int kb2 = ks2 * 32;
#pragma unroll
            for (int mt = 0; mt < 4; ++mt) {
                v16bf a = load_frag(hA, HA_S, mt * 16 + l16, kb2 + lhi * 8);
#pragma unroll
                for (int nt = 0; nt < 4; ++nt) {
                    int n = w * 64 + nt * 16 + l16;
                    v16bf b = load_frag(B2, B2_S, n, kb2 + lhi * 8);
                    C2[mt * 4 + nt] = __builtin_amdgcn_wmma_f32_16x16x32_bf16(
                        false, a, false, b, (short)0, C2[mt * 4 + nt], false, false);
                }
            }
        }
    }

    // Weighted scatter-add into output (+ b2)
#pragma unroll
    for (int mt = 0; mt < 4; ++mt) {
#pragma unroll
        for (int r = 0; r < 8; ++r) {
            int m = mt * 16 + lhi * 8 + r;
            if (m < rowsValid) {
                int tok = tokList[m];
                float pw = wList[m];
#pragma unroll
                for (int nt = 0; nt < 4; ++nt) {
                    int n = w * 64 + nt * 16 + l16;
                    float val = pw * (C2[mt * 4 + nt][r] + b2[(size_t)e * DIM + n]);
                    atomicAdd(out + (size_t)tok * DIM + n, val);
                }
            }
        }
    }
}

// ---------------- host-side launch ----------------
extern "C" void kernel_launch(void* const* d_in, const int* in_sizes, int n_in,
                              void* d_out, int out_size, void* d_ws, size_t ws_size,
                              hipStream_t stream) {
    const float* x  = (const float*)d_in[0];
    const float* gW = (const float*)d_in[1];
    const float* gb = (const float*)d_in[2];
    const float* W1 = (const float*)d_in[3];
    const float* b1 = (const float*)d_in[4];
    const float* W2 = (const float*)d_in[5];
    const float* b2 = (const float*)d_in[6];
    float* out = (float*)d_out;

    char* wsb = (char*)d_ws;
    int*   cnt     = (int*)(wsb + 0);     // 8 ints
    int*   cur     = (int*)(wsb + 64);    // 8 ints
    float* ssum    = (float*)(wsb + 128); // 8 floats
    int*   e0      = (int*)(wsb + 256);
    int*   e1      = e0 + T_TOK;
    float* w0      = (float*)(e1 + T_TOK);
    float* w1      = w0 + T_TOK;
    int*   pairTok = (int*)(w1 + T_TOK);
    float* pairW   = (float*)(pairTok + (size_t)NEXP * T_TOK);

    const int nOut = T_TOK * DIM + 1;
    moe_zero_kernel<<<(nOut + 255) / 256, 256, 0, stream>>>(out, nOut, (unsigned int*)wsb);

    moe_gate_kernel<<<T_TOK / 8, 256, 0, stream>>>(x, gW, gb, e0, e1, w0, w1, cnt, ssum);

    moe_scatter_kernel<<<(T_TOK + 255) / 256, 256, 0, stream>>>(e0, e1, w0, w1, cur,
                                                               pairTok, pairW);

    moe_aux_kernel<<<1, 64, 0, stream>>>(cnt, ssum, out + (size_t)T_TOK * DIM);

    dim3 grid(128, NEXP);
    moe_ffn_kernel<<<grid, 512, LDS_BYTES, stream>>>(x, W1, b1, W2, b2, pairTok, pairW,
                                                     cnt, out);
}
